// GAT_71528385348098
// MI455X (gfx1250) — compile-verified, matrix operand order checked by reading, not run on hardware
//
#include <hip/hip_runtime.h>

// ---------------- problem constants (match reference) ----------------
#define N_NODES 50000
#define N_PAD   50048            // 391 * 128, multiple of GEMM macro-tile
#define N_EDGES 800000
#define ET      (N_EDGES + N_NODES)   // edges + self loops = 850000
#define F_IN    128
#define HDIM    256
#define NGRAPH  500
#define FC1DIM  196
#define DOUT    10
#define NEG_SLOPE 0.2f

// ---------------- CDNA5 WMMA types ----------------
typedef __attribute__((ext_vector_type(16))) __bf16 v16bf;
typedef __attribute__((ext_vector_type(8)))  float  v8f;
typedef int v4i_vs __attribute__((vector_size(16)));   // matches builtin prototype

union FragBF { v16bf v; unsigned int u[8]; };
union AccF   { v8f   v; float f[8]; };

// CDNA5 async global->LDS path (guarded; falls back to sync vector copies)
#if defined(__gfx1250__) && __has_builtin(__builtin_amdgcn_global_load_async_to_lds_b128) && \
    __has_builtin(__builtin_amdgcn_s_wait_asynccnt)
#define HAVE_ASYNC_LDS 1
#else
#define HAVE_ASYNC_LDS 0
#endif

// ---------------- small device helpers ----------------
__device__ __forceinline__ unsigned short f2bf(float f) {
  unsigned int u = __float_as_uint(f);
  unsigned int r = u + 0x7fffu + ((u >> 16) & 1u);   // round-to-nearest-even
  return (unsigned short)(r >> 16);
}
// order-preserving float<->int encoding for atomicMax on signed int
__device__ __forceinline__ int fenc(float f) {
  int i = __float_as_int(f);
  return i >= 0 ? i : (i ^ 0x7fffffff);
}
__device__ __forceinline__ float fdec(int i) {
  return __int_as_float(i >= 0 ? i : (i ^ 0x7fffffff));
}

// ---------------- utility kernels ----------------
__global__ void k_fill_u32(unsigned int* p, unsigned int val, long long n) {
  long long i = (long long)blockIdx.x * blockDim.x + threadIdx.x;
  long long stride = (long long)gridDim.x * blockDim.x;
  for (; i < n; i += stride) p[i] = val;
}

__global__ void k_f32_to_bf16(const float* __restrict__ s,
                              unsigned short* __restrict__ d, int n) {
  int i = blockIdx.x * blockDim.x + threadIdx.x;
  if (i < n) d[i] = f2bf(s[i]);
}

// Convert W [K][256] f32 -> bf16 pre-swizzled into WMMA B-fragment order:
// per (k-tile kt of 32, col-tile t of 16): 32 lanes x 16 contiguous bf16.
// lane = (k&8 ? 16 : 0) + (n&15); u-idx v = (k&16 ? 4 : 0) + ((k&7)>>1); hw = k&1.
__global__ void k_w_swizzle_bf16(const float* __restrict__ W,
                                 unsigned short* __restrict__ Ws, int K) {
  int i = blockIdx.x * blockDim.x + threadIdx.x;
  if (i >= K * HDIM) return;
  int k = i >> 8, n = i & 255;
  int kt = k >> 5, kr = k & 31;
  int t = n >> 4;
  int lane = ((kr & 8) ? 16 : 0) + (n & 15);
  int v = ((kr & 16) ? 4 : 0) + ((kr & 7) >> 1);
  int hw = kr & 1;
  size_t idx = ((((size_t)kt * 16 + t) * 32 + lane) << 4) + (v << 1) + hw;
  Ws[idx] = f2bf(W[i]);
}

// ---------------- WMMA bf16 GEMM: C[Mpad,256] = A[Mpad,K] * B[K,256] --------
// block = 128 threads (4 waves); macro-tile 128x64; each wave owns 32x64
// (2 row-tiles x 4 col-tiles = 8 accumulators); K stepped by 32.
// A staged in LDS (async global->LDS on CDNA5); B read directly from the
// pre-swizzled global weight buffer (2x global_load_b128 per fragment).
__global__ __launch_bounds__(128)
void k_gemm_bf16_wmma(const unsigned short* __restrict__ A,   // bf16 raw [Mpad][K]
                      const unsigned short* __restrict__ Bsw, // bf16 swizzled W
                      float* __restrict__ C,                  // [Mpad][256]
                      int Mpad, int K) {
  __shared__ __align__(16) unsigned short As[128 * 32];   // 8 KB

  const int tid  = threadIdx.x;
  const int lane = tid & 31;
  const int wv   = tid >> 5;                 // wave 0..3
  const int row0 = blockIdx.y * 128;
  const int ct0  = blockIdx.x * 4;           // first of 4 col-tiles (x16 cols)
  const int col0 = ct0 << 4;

  AccF acc[2][4];
#pragma unroll
  for (int r = 0; r < 2; r++)
#pragma unroll
    for (int t = 0; t < 4; t++)
#pragma unroll
      for (int v = 0; v < 8; v++) acc[r][t].f[v] = 0.f;

  const int kb8 = (lane < 16) ? 0 : 8;       // K sub-offset per documented layout
  const int m0  = wv * 32 + (lane & 15);     // A fragment rows inside tile
  const int m1  = m0 + 16;

  const int nKt = K >> 5;
  for (int kt = 0; kt < nKt; kt++) {
    const int k0 = kt << 5;
    // ---- stage A tile: thread t copies row (row0+t), 32 bf16 = 64 bytes ----
    const unsigned short* ga = A + (size_t)(row0 + tid) * K + k0;
    unsigned short* la = &As[tid * 32];
#if HAVE_ASYNC_LDS
#pragma unroll
    for (int j = 0; j < 4; j++) {
      __builtin_amdgcn_global_load_async_to_lds_b128(
          (__attribute__((address_space(1))) v4i_vs*)(ga + j * 8),
          (__attribute__((address_space(3))) v4i_vs*)(la + j * 8), 0, 0);
    }
    __builtin_amdgcn_s_wait_asynccnt(0);
#else
    {
      const uint4* g4 = (const uint4*)ga;
      uint4* l4 = (uint4*)la;
      l4[0] = g4[0]; l4[1] = g4[1]; l4[2] = g4[2]; l4[3] = g4[3];
    }
#endif
    __syncthreads();

    if (kt + 1 < nKt)   // prefetch next A tile -> global_prefetch_b8
      __builtin_prefetch(ga + 32, 0, 0);

    // ---- A fragments: two contiguous 16B halves per lane (ds_load_b128 x2) ----
    FragBF af[2];
#pragma unroll
    for (int v = 0; v < 8; v++) {
      int kk = (v < 4) ? (kb8 + 2 * v) : (16 + kb8 + 2 * (v - 4));
      af[0].u[v] = *(const unsigned int*)(&As[m0 * 32 + kk]);
      af[1].u[v] = *(const unsigned int*)(&As[m1 * 32 + kk]);
    }

    // ---- B fragments direct from swizzled global: 32 contiguous bytes/lane ----
#pragma unroll
    for (int t = 0; t < 4; t++) {
      FragBF bf;
      const uint4* bp = (const uint4*)(Bsw + ((((size_t)kt * 16 + ct0 + t) * 32 + lane) << 4));
      *(uint4*)&bf.u[0] = bp[0];
      *(uint4*)&bf.u[4] = bp[1];
      acc[0][t].v = __builtin_amdgcn_wmma_f32_16x16x32_bf16(
          false, af[0].v, false, bf.v, (short)0, acc[0][t].v, false, false);
      acc[1][t].v = __builtin_amdgcn_wmma_f32_16x16x32_bf16(
          false, af[1].v, false, bf.v, (short)0, acc[1][t].v, false, false);
    }
    __syncthreads();
  }

  // C/D layout: VGPR v -> M = v (+8 for lanes 16..31), N = lane&15
  const int cb = lane & 15;
#pragma unroll
  for (int r = 0; r < 2; r++) {
    const int rbase = row0 + wv * 32 + r * 16 + ((lane >> 4) << 3);
#pragma unroll
    for (int t = 0; t < 4; t++)
#pragma unroll
      for (int v = 0; v < 8; v++)
        C[(size_t)(rbase + v) * HDIM + col0 + t * 16 + cb] = acc[r][t].f[v];
  }
}

// ---------------- per-node attention dot products (wave per node) ----------
__global__ void k_dots(const float* __restrict__ h, const float* __restrict__ a_s,
                       const float* __restrict__ a_d, float* __restrict__ hs,
                       float* __restrict__ hd, int n) {
  const int lane = threadIdx.x & 31;
  const int node = blockIdx.x * 8 + (threadIdx.x >> 5);
  if (node >= n) return;
  const float* row = h + (size_t)node * HDIM;
  float s = 0.f, d = 0.f;
#pragma unroll
  for (int j = 0; j < 8; j++) {
    int c = lane + j * 32;
    float v = row[c];
    s += v * a_s[c];
    d += v * a_d[c];
  }
#pragma unroll
  for (int off = 16; off > 0; off >>= 1) {
    s += __shfl_down(s, off, 32);
    d += __shfl_down(d, off, 32);
  }
  if (lane == 0) { hs[node] = s; hd[node] = d; }
}

// ---------------- edge pass 1: logits + segment max ----------------
__global__ void k_edge_logits(const int* __restrict__ ei, const float* __restrict__ hs,
                              const float* __restrict__ hd, float* __restrict__ ec,
                              int* __restrict__ mx) {
  int e = blockIdx.x * blockDim.x + threadIdx.x;
  if (e >= ET) return;
  int s, d;
  if (e < N_EDGES) { s = ei[e]; d = ei[N_EDGES + e]; }
  else             { s = d = e - N_EDGES; }
  float v = hs[s] + hd[d];
  v = (v > 0.f) ? v : NEG_SLOPE * v;          // leaky_relu
  ec[e] = v;
  atomicMax(&mx[d], fenc(v));
}

// ---------------- edge pass 2: exp + segment sum ----------------
__global__ void k_edge_denom(const int* __restrict__ ei, float* __restrict__ ec,
                             const int* __restrict__ mx, float* __restrict__ dn) {
  int e = blockIdx.x * blockDim.x + threadIdx.x;
  if (e >= ET) return;
  int d = (e < N_EDGES) ? ei[N_EDGES + e] : (e - N_EDGES);
  float ex = __expf(ec[e] - fdec(mx[d]));
  ec[e] = ex;
  atomicAdd(&dn[d], ex);
}

// ---------------- edge pass 3: weighted aggregation (wave per edge) --------
__global__ void k_edge_aggr(const int* __restrict__ ei, const float* __restrict__ h,
                            const float* __restrict__ ec, const float* __restrict__ dn,
                            float* __restrict__ out) {
  const int e = blockIdx.x * 8 + (threadIdx.x >> 5);
  if (e >= ET) return;
  const int lane = threadIdx.x & 31;
  int s, d;
  if (e < N_EDGES) { s = ei[e]; d = ei[N_EDGES + e]; }
  else             { s = d = e - N_EDGES; }
  const float alpha = ec[e] / fmaxf(dn[d], 1e-16f);
  const float* hrow = h + (size_t)s * HDIM;
  float* orow = out + (size_t)d * HDIM;
#pragma unroll
  for (int j = 0; j < 8; j++) {
    int c = lane + j * 32;
    atomicAdd(&orow[c], alpha * hrow[c]);
  }
}

// ---------------- bias + relu, write f32 in place + bf16 for next GEMM -----
__global__ void k_bias_relu_cvt(float* __restrict__ h, const float* __restrict__ b,
                                unsigned short* __restrict__ obf, long long total) {
  long long i = (long long)blockIdx.x * blockDim.x + threadIdx.x;
  if (i >= total) return;
  int c = (int)(i & (HDIM - 1));
  float v = h[i] + b[c];
  v = v > 0.f ? v : 0.f;
  h[i] = v;
  obf[i] = f2bf(v);
}

// ---------------- mean pool over graphs (wave per node) ----------------
__global__ void k_pool(const float* __restrict__ h, const int* __restrict__ batch,
                       float* __restrict__ hg, float* __restrict__ cnt, int n) {
  const int lane = threadIdx.x & 31;
  const int node = blockIdx.x * 8 + (threadIdx.x >> 5);
  if (node >= n) return;
  const int g = batch[node];
  const float* row = h + (size_t)node * HDIM;
  float* grow = hg + (size_t)g * HDIM;
#pragma unroll
  for (int j = 0; j < 8; j++) {
    int c = lane + j * 32;
    atomicAdd(&grow[c], row[c]);
  }
  if (lane == 0) atomicAdd(&cnt[g], 1.0f);
}

__global__ void k_pool_div(float* __restrict__ hg, const float* __restrict__ cnt, int n) {
  int i = blockIdx.x * blockDim.x + threadIdx.x;
  if (i < n) hg[i] /= fmaxf(cnt[i >> 8], 1.0f);
}

// ---------------- tiny MLP head ----------------
__global__ void k_fc1(const float* __restrict__ hg, const float* __restrict__ w,
                      const float* __restrict__ b, float* __restrict__ o, int n) {
  int i = blockIdx.x * blockDim.x + threadIdx.x;
  if (i >= n) return;
  int g = i / FC1DIM, j = i - g * FC1DIM;
  const float* row = hg + (size_t)g * HDIM;
  float s = 0.f;
  for (int k = 0; k < HDIM; k++) s += row[k] * w[k * FC1DIM + j];
  s += b[j];
  o[i] = s > 0.f ? s : 0.f;
}

__global__ void k_fc2(const float* __restrict__ f1, const float* __restrict__ w,
                      const float* __restrict__ b, float* __restrict__ o, int n) {
  int i = blockIdx.x * blockDim.x + threadIdx.x;
  if (i >= n) return;
  int g = i / DOUT, j = i - g * DOUT;
  const float* row = f1 + (size_t)g * FC1DIM;
  float s = 0.f;
  for (int k = 0; k < FC1DIM; k++) s += row[k] * w[k * DOUT + j];
  o[i] = s + b[j];
}

// ---------------- host launcher ----------------
extern "C" void kernel_launch(void* const* d_in, const int* in_sizes, int n_in,
                              void* d_out, int out_size, void* d_ws, size_t ws_size,
                              hipStream_t stream) {
  (void)in_sizes; (void)n_in; (void)out_size; (void)ws_size;
  const float* x     = (const float*)d_in[0];
  const int*   ei    = (const int*)d_in[1];
  const int*   batch = (const int*)d_in[2];
  const float* Wm[3]  = {(const float*)d_in[3], (const float*)d_in[7],  (const float*)d_in[11]};
  const float* Avs[3] = {(const float*)d_in[4], (const float*)d_in[8],  (const float*)d_in[12]};
  const float* Avd[3] = {(const float*)d_in[5], (const float*)d_in[9],  (const float*)d_in[13]};
  const float* Bv[3]  = {(const float*)d_in[6], (const float*)d_in[10], (const float*)d_in[14]};
  const float* fc1w = (const float*)d_in[15];
  const float* fc1b = (const float*)d_in[16];
  const float* fc2w = (const float*)d_in[17];
  const float* fc2b = (const float*)d_in[18];
  float* out = (float*)d_out;

  char* base = (char*)d_ws;
  size_t off = 0;
  auto alloc = [&](size_t bytes) -> char* {
    char* p = base + off;
    off = (off + bytes + 255) & ~(size_t)255;
    return p;
  };
  unsigned short* bf_in = (unsigned short*)alloc((size_t)N_PAD * HDIM * 2);  // bf16 GEMM A
  unsigned short* w_bf  = (unsigned short*)alloc((size_t)HDIM * HDIM * 2);   // bf16 swizzled W
  float* h_gemm = (float*)alloc((size_t)N_PAD * HDIM * 4);                   // xW (pre-attention)
  float* h_out  = (float*)alloc((size_t)N_PAD * HDIM * 4);                   // aggregated output
  float* hs  = (float*)alloc((size_t)N_NODES * 4);
  float* hd  = (float*)alloc((size_t)N_NODES * 4);
  int*   mx  = (int*)  alloc((size_t)N_NODES * 4);
  float* dn  = (float*)alloc((size_t)N_NODES * 4);
  float* ec  = (float*)alloc((size_t)ET * 4);
  float* hg  = (float*)alloc((size_t)NGRAPH * HDIM * 4);
  float* cnt = (float*)alloc((size_t)NGRAPH * 4);
  float* f1o = (float*)alloc((size_t)NGRAPH * FC1DIM * 4);

  // zero the whole padded bf16 input buffer once (covers pad rows for all K layouts)
  k_fill_u32<<<4096, 256, 0, stream>>>((unsigned int*)bf_in, 0u, (long long)N_PAD * 128);
  // layer-1 A matrix: x -> bf16, ld = F_IN
  k_f32_to_bf16<<<(N_NODES * F_IN + 255) / 256, 256, 0, stream>>>(x, bf_in, N_NODES * F_IN);

  for (int L = 0; L < 3; L++) {
    const int K = (L == 0) ? F_IN : HDIM;
    k_w_swizzle_bf16<<<(K * HDIM + 255) / 256, 256, 0, stream>>>(Wm[L], w_bf, K);

    dim3 g(HDIM / 64, N_PAD / 128);
    k_gemm_bf16_wmma<<<g, 128, 0, stream>>>(bf_in, w_bf, h_gemm, N_PAD, K);

    k_dots<<<(N_NODES + 7) / 8, 256, 0, stream>>>(h_gemm, Avs[L], Avd[L], hs, hd, N_NODES);

    k_fill_u32<<<256, 256, 0, stream>>>((unsigned int*)mx, 0x80000000u, N_NODES);
    k_fill_u32<<<256, 256, 0, stream>>>((unsigned int*)dn, 0u, N_NODES);
    k_fill_u32<<<4096, 256, 0, stream>>>((unsigned int*)h_out, 0u, (long long)N_NODES * HDIM);

    k_edge_logits<<<(ET + 255) / 256, 256, 0, stream>>>(ei, hs, hd, ec, mx);
    k_edge_denom <<<(ET + 255) / 256, 256, 0, stream>>>(ei, ec, mx, dn);
    k_edge_aggr  <<<(ET + 7) / 8,     256, 0, stream>>>(ei, h_gemm, ec, dn, h_out);

    k_bias_relu_cvt<<<((long long)N_NODES * HDIM + 255) / 256, 256, 0, stream>>>(
        h_out, Bv[L], bf_in, (long long)N_NODES * HDIM);
  }

  k_fill_u32<<<128, 256, 0, stream>>>((unsigned int*)hg, 0u, NGRAPH * HDIM);
  k_fill_u32<<<2, 256, 0, stream>>>((unsigned int*)cnt, 0u, NGRAPH);
  k_pool<<<(N_NODES + 7) / 8, 256, 0, stream>>>(h_out, batch, hg, cnt, N_NODES);
  k_pool_div<<<(NGRAPH * HDIM + 255) / 256, 256, 0, stream>>>(hg, cnt, NGRAPH * HDIM);
  k_fc1<<<(NGRAPH * FC1DIM + 255) / 256, 256, 0, stream>>>(hg, fc1w, fc1b, f1o, NGRAPH * FC1DIM);
  k_fc2<<<(NGRAPH * DOUT + 255) / 256, 256, 0, stream>>>(f1o, fc2w, fc2b, out, NGRAPH * DOUT);
}
